// TrackGraphRegressorGNN_7799660609778
// MI455X (gfx1250) — matrix operand order repaired and checked by reading, not compiled, not run on hardware
//
#include <hip/hip_runtime.h>
#include <hip/hip_bf16.h>

// ---------------------------------------------------------------------------
// GNN layer for MI455X (gfx1250, wave32, WMMA).
//   pre-pack : f32 weights -> f16 WMMA B-fragment layout (once per launch)
//   edge MLP : [E,136] @ [136,128] -> ReLU -> @ [128,64]   (v_wmma f16->f32)
//   scatter  : global_atomic_add_f32 into agg[N,64] (L2-resident, 12.8MB)
//   node MLP : [N,128] @ [128,128] -> ReLU -> @ [128,64] + residual + LN
// ---------------------------------------------------------------------------

typedef __attribute__((ext_vector_type(16))) _Float16 v16h;
typedef __attribute__((ext_vector_type(8)))  float    v8f;

#define HDIM  64
#define EDIM  8
#define MSGH  128
#define K1P   160   // 2*HDIM + EDIM (=136) padded to a multiple of 32
#define SAP   168   // sA row stride (halves): 160 + 8 pad (anti bank-conflict)
#define SHP   136   // sH row stride (halves): 128 + 8 pad
#define SNP   136   // node sA row stride
#define SOP   65    // sO row stride (floats): 64 + 1 pad
#define TILE  64    // rows (edges / nodes) per tile
#define NTHR  256   // 8 wave32 per block

// packed-weight fragment counts (frag = 32 lanes x 16 halves = 512 halves)
#define F_EW1 (5 * 8)   // K=160/32 chunks x 128/16 col-tiles
#define F_EW2 (4 * 4)
#define F_NW1 (4 * 8)
#define F_NW2 (4 * 4)

__device__ __forceinline__ v8f wmma_f16(v16h a, v16h b, v8f c) {
    return __builtin_amdgcn_wmma_f32_16x16x32_f16(
        false, a, false, b, (short)0, c, false, false);
}

// A fragment (16x32 f16) from LDS: lane holds row M=lane%16; two b128 loads
// at K-offsets cb and 8+cb, cb = (lane>=16)*16.  (contiguous k order; B packed to match)
__device__ __forceinline__ v16h load_a_frag(const _Float16* sA, int rowHalfs,
                                            int mbase, int k0, int lane) {
    int row = mbase + (lane & 15);
    int cb  = (lane >> 4) << 4;
    const _Float16* p = sA + row * rowHalfs + k0 + cb;
    union { v16h v; uint4 u[2]; } r;
    r.u[0] = *(const uint4*)p;
    r.u[1] = *(const uint4*)(p + 8);
    return r.v;
}

// B fragment (32x16 f16) from the pre-packed buffer: 32 bytes per lane,
// two global_load_b128, no conversion.
__device__ __forceinline__ v16h load_b_packed(const _Float16* __restrict__ P,
                                              int ntiles, int kc, int ct,
                                              int lane) {
    const _Float16* p = P + (((size_t)(kc * ntiles + ct) * 32) + lane) * 16;
    union { v16h v; uint4 u[2]; } r;
    r.u[0] = *(const uint4*)p;
    r.u[1] = *(const uint4*)(p + 8);
    return r.v;
}

// ---------------------------------------------------------------------------
__global__ void zero_f32_kernel(float* __restrict__ p, size_t n) {
    size_t i = (size_t)blockIdx.x * blockDim.x + threadIdx.x;
    if (i < n) p[i] = 0.0f;
}

// Pack W[kmax x ncols] (f32, row-major) into WMMA B-fragment layout (f16).
// Block = one fragment (kc, col-tile); thread = lane. Lane layout:
// column N = lane%16, halves[j] = W[kc*32 + (lane>=16)*16 + j][col].
__global__ __launch_bounds__(32) void
pack_weights_kernel(const float* __restrict__ W, _Float16* __restrict__ P,
                    int kmax, int ncols) {
    const int f      = blockIdx.x;
    const int lane   = threadIdx.x;
    const int ntiles = ncols >> 4;
    const int kc     = f / ntiles;
    const int ct     = f % ntiles;
    const int col    = ct * 16 + (lane & 15);
    const int kb     = kc * 32 + ((lane >> 4) << 4);
    _Float16* dst = P + ((size_t)f * 32 + lane) * 16;
#pragma unroll
    for (int j = 0; j < 16; ++j) {
        int k = kb + j;
        dst[j] = (k < kmax) ? (_Float16)W[(size_t)k * ncols + col]
                            : (_Float16)0.0f;
    }
}

// ---------------------------------------------------------------------------
// Edge kernel: gather -> edge MLP (two WMMA GEMMs) -> atomic scatter-add.
// Grid-stride over 64-edge tiles; B fragments stay resident in VGPRs.
// ---------------------------------------------------------------------------
__global__ __launch_bounds__(NTHR) void
edge_mlp_scatter_kernel(const float* __restrict__ h,
                        const int*   __restrict__ ei,    // [2,E]
                        const float* __restrict__ ea,    // [E,EDIM]
                        const _Float16* __restrict__ eW1p,
                        const float* __restrict__ eb1,   // [128]
                        const _Float16* __restrict__ eW2p,
                        const float* __restrict__ eb2,   // [64]
                        float* __restrict__ agg,         // [N,64]
                        int E) {
    __shared__ _Float16 sA[TILE * SAP];   // gathered edge inputs (f16)
    __shared__ _Float16 sH[TILE * SHP];   // hidden activations (f16)
    __shared__ int      sDst[TILE];

    const int t    = threadIdx.x;
    const int lane = t & 31;
    const int wv   = t >> 5;          // wave id 0..7

    // ---- resident B fragments (pre-packed f16), 2 b128 loads each ----
    const int colW1 = wv * 16 + (lane & 15);          // hidden column
    v16h b1[5];
#pragma unroll
    for (int kc = 0; kc < 5; ++kc)
        b1[kc] = load_b_packed(eW1p, 8, kc, wv, lane);

    const int n2base = (wv & 1) * 2;                  // GEMM2 n-tile pair
    v16h b2[2][4];
#pragma unroll
    for (int j = 0; j < 2; ++j)
#pragma unroll
        for (int kc = 0; kc < 4; ++kc)
            b2[j][kc] = load_b_packed(eW2p, 4, kc, n2base + j, lane);

    const float bias1 = eb1[colW1];
    const int   mt2   = wv >> 1;
    const int   le    = t >> 2;       // local edge 0..63 (gather)
    const int   part  = t & 3;        // 32-float slice

    const int ntile = (E + TILE - 1) / TILE;
    for (int tile = blockIdx.x; tile < ntile; tile += gridDim.x) {
        const int e0 = tile * TILE;

        // ---- gather: 4 threads per edge, f32 -> f16 into LDS A tile ----
        {
            const int eg = e0 + le;
            _Float16* dptr = sA + le * SAP + part * 32;
            if (eg < E) {
                const int src = ei[eg];
                const int dst = ei[E + eg];
                const float* hp = ((part < 2) ? (h + (size_t)src * HDIM)
                                              : (h + (size_t)dst * HDIM))
                                  + (part & 1) * 32;
#pragma unroll
                for (int i = 0; i < 32; i += 4) {
                    float4 v = *(const float4*)(hp + i);
                    dptr[i + 0] = (_Float16)v.x; dptr[i + 1] = (_Float16)v.y;
                    dptr[i + 2] = (_Float16)v.z; dptr[i + 3] = (_Float16)v.w;
                }
                if (part == 0) {
                    sDst[le] = dst;
                    float4 a0 = *(const float4*)(ea + (size_t)eg * EDIM);
                    float4 a1 = *(const float4*)(ea + (size_t)eg * EDIM + 4);
                    _Float16* ap = sA + le * SAP + 2 * HDIM;
                    ap[0] = (_Float16)a0.x; ap[1] = (_Float16)a0.y;
                    ap[2] = (_Float16)a0.z; ap[3] = (_Float16)a0.w;
                    ap[4] = (_Float16)a1.x; ap[5] = (_Float16)a1.y;
                    ap[6] = (_Float16)a1.z; ap[7] = (_Float16)a1.w;
#pragma unroll
                    for (int i = EDIM; i < K1P - 2 * HDIM; ++i)
                        ap[i] = (_Float16)0.0f;   // K padding 136..159
                }
            } else {
#pragma unroll
                for (int i = 0; i < 32; ++i) dptr[i] = (_Float16)0.0f;
                if (part == 0) {
                    sDst[le] = 0;
                    _Float16* ap = sA + le * SAP + 2 * HDIM;
#pragma unroll
                    for (int i = 0; i < K1P - 2 * HDIM; ++i)
                        ap[i] = (_Float16)0.0f;
                }
            }
        }
        __syncthreads();

        // ---- GEMM1: [64x160] x [160x128] -> ReLU -> sH ----
#pragma unroll
        for (int mt = 0; mt < 4; ++mt) {
            v8f c;
#pragma unroll
            for (int i = 0; i < 8; ++i) c[i] = bias1;
#pragma unroll
            for (int kc = 0; kc < 5; ++kc) {
                v16h a = load_a_frag(sA, SAP, mt * 16, kc * 32, lane);
                c = wmma_f16(a, b1[kc], c);
            }
#pragma unroll
            for (int i = 0; i < 8; ++i) {
                int M = mt * 16 + i + ((lane >> 4) << 3);
                sH[M * SHP + colW1] = (_Float16)fmaxf(c[i], 0.0f);
            }
        }
        __syncthreads();

        // ---- GEMM2: [64x128] x [128x64] + eb2, atomic scatter by dst ----
#pragma unroll
        for (int j = 0; j < 2; ++j) {
            const int colg = (n2base + j) * 16 + (lane & 15);
            const float bias2 = eb2[colg];
            v8f c;
#pragma unroll
            for (int i = 0; i < 8; ++i) c[i] = bias2;
#pragma unroll
            for (int kc = 0; kc < 4; ++kc) {
                v16h a = load_a_frag(sH, SHP, mt2 * 16, kc * 32, lane);
                c = wmma_f16(a, b2[j][kc], c);
            }
#pragma unroll
            for (int i = 0; i < 8; ++i) {
                int M = mt2 * 16 + i + ((lane >> 4) << 3);
                if (e0 + M < E) {
                    int d = sDst[M];
                    atomicAdd(agg + (size_t)d * HDIM + colg, c[i]);
                }
            }
        }
        __syncthreads();   // protect sA/sH before next tile overwrites
    }
}

// ---------------------------------------------------------------------------
// Node kernel: [h|agg] -> node MLP (two WMMA GEMMs) -> residual -> LayerNorm.
// ---------------------------------------------------------------------------
__global__ __launch_bounds__(NTHR) void
node_mlp_ln_kernel(const float* __restrict__ h,
                   const float* __restrict__ agg,
                   const _Float16* __restrict__ nW1p,
                   const float* __restrict__ nb1,  // [128]
                   const _Float16* __restrict__ nW2p,
                   const float* __restrict__ nb2,  // [64]
                   const float* __restrict__ lng,  // [64]
                   const float* __restrict__ lnb,  // [64]
                   float* __restrict__ out, int N) {
    __shared__ _Float16 sA[TILE * SNP];
    __shared__ _Float16 sH[TILE * SHP];
    __shared__ float    sO[TILE * SOP];

    const int t    = threadIdx.x;
    const int lane = t & 31;
    const int wv   = t >> 5;
    const int r0   = blockIdx.x * TILE;

    const int colW1 = wv * 16 + (lane & 15);
    v16h b1[4];
#pragma unroll
    for (int kc = 0; kc < 4; ++kc)
        b1[kc] = load_b_packed(nW1p, 8, kc, wv, lane);

    const int n2base = (wv & 1) * 2;
    v16h b2[2][4];
#pragma unroll
    for (int j = 0; j < 2; ++j)
#pragma unroll
        for (int kc = 0; kc < 4; ++kc)
            b2[j][kc] = load_b_packed(nW2p, 4, kc, n2base + j, lane);

    // ---- load A tile: [h | agg] -> f16 ----
    {
        const int lr   = t >> 2;
        const int part = t & 3;
        const int gr   = r0 + lr;
        _Float16* dptr = sA + lr * SNP + part * 32;
        if (gr < N) {
            const float* sp = (part < 2)
                ? (h   + (size_t)gr * HDIM + part * 32)
                : (agg + (size_t)gr * HDIM + (part - 2) * 32);
#pragma unroll
            for (int i = 0; i < 32; i += 4) {
                float4 v = *(const float4*)(sp + i);
                dptr[i + 0] = (_Float16)v.x; dptr[i + 1] = (_Float16)v.y;
                dptr[i + 2] = (_Float16)v.z; dptr[i + 3] = (_Float16)v.w;
            }
        } else {
#pragma unroll
            for (int i = 0; i < 32; ++i) dptr[i] = (_Float16)0.0f;
        }
    }
    __syncthreads();

    // ---- GEMM1 -> ReLU -> sH ----
    const float bias1 = nb1[colW1];
#pragma unroll
    for (int mt = 0; mt < 4; ++mt) {
        v8f c;
#pragma unroll
        for (int i = 0; i < 8; ++i) c[i] = bias1;
#pragma unroll
        for (int kc = 0; kc < 4; ++kc) {
            v16h a = load_a_frag(sA, SNP, mt * 16, kc * 32, lane);
            c = wmma_f16(a, b1[kc], c);
        }
#pragma unroll
        for (int i = 0; i < 8; ++i) {
            int M = mt * 16 + i + ((lane >> 4) << 3);
            sH[M * SHP + colW1] = (_Float16)fmaxf(c[i], 0.0f);
        }
    }
    __syncthreads();

    // ---- GEMM2 + bias + residual -> sO (f32) ----
    const int mt2 = wv >> 1;
#pragma unroll
    for (int j = 0; j < 2; ++j) {
        const int colg = (n2base + j) * 16 + (lane & 15);
        const float bias2 = nb2[colg];
        v8f c;
#pragma unroll
        for (int i = 0; i < 8; ++i) c[i] = bias2;
#pragma unroll
        for (int kc = 0; kc < 4; ++kc) {
            v16h a = load_a_frag(sH, SHP, mt2 * 16, kc * 32, lane);
            c = wmma_f16(a, b2[j][kc], c);
        }
#pragma unroll
        for (int i = 0; i < 8; ++i) {
            int M   = mt2 * 16 + i + ((lane >> 4) << 3);
            int gr  = r0 + M;
            float res = (gr < N) ? h[(size_t)gr * HDIM + colg] : 0.0f;
            sO[M * SOP + colg] = c[i] + res;
        }
    }
    __syncthreads();

    // ---- LayerNorm over 64 features, one row per thread ----
    if (t < TILE) {
        const int gr = r0 + t;
        if (gr < N) {
            const float* row = sO + t * SOP;
            float mu = 0.0f;
#pragma unroll
            for (int c = 0; c < HDIM; ++c) mu += row[c];
            mu *= (1.0f / HDIM);
            float var = 0.0f;
#pragma unroll
            for (int c = 0; c < HDIM; ++c) {
                float d = row[c] - mu;
                var += d * d;
            }
            var *= (1.0f / HDIM);
            const float rs = rsqrtf(var + 1e-5f);
#pragma unroll
            for (int c = 0; c < HDIM; ++c)
                out[(size_t)gr * HDIM + c] =
                    (row[c] - mu) * rs * lng[c] + lnb[c];
        }
    }
}

// ---------------------------------------------------------------------------
extern "C" void kernel_launch(void* const* d_in, const int* in_sizes, int n_in,
                              void* d_out, int out_size, void* d_ws, size_t ws_size,
                              hipStream_t stream) {
    const float* h   = (const float*)d_in[0];
    const int*   ei  = (const int*)  d_in[1];
    const float* ea  = (const float*)d_in[2];
    const float* eW1 = (const float*)d_in[3];
    const float* eb1 = (const float*)d_in[4];
    const float* eW2 = (const float*)d_in[5];
    const float* eb2 = (const float*)d_in[6];
    const float* nW1 = (const float*)d_in[7];
    const float* nb1 = (const float*)d_in[8];
    const float* nW2 = (const float*)d_in[9];
    const float* nb2 = (const float*)d_in[10];
    const float* lng = (const float*)d_in[11];
    const float* lnb = (const float*)d_in[12];

    const int N = in_sizes[0] / HDIM;
    const int E = in_sizes[1] / 2;

    // workspace layout: agg[N*64] f32, then packed f16 weights
    float*    agg  = (float*)d_ws;
    const size_t nAgg = (size_t)N * HDIM;
    _Float16* eW1p = (_Float16*)(agg + nAgg);
    _Float16* eW2p = eW1p + (size_t)F_EW1 * 512;
    _Float16* nW1p = eW2p + (size_t)F_EW2 * 512;
    _Float16* nW2p = nW1p + (size_t)F_NW1 * 512;

    zero_f32_kernel<<<(unsigned)((nAgg + 1023) / 1024), 1024, 0, stream>>>(
        agg, nAgg);

    pack_weights_kernel<<<F_EW1, 32, 0, stream>>>(eW1, eW1p, 2 * HDIM + EDIM, MSGH);
    pack_weights_kernel<<<F_EW2, 32, 0, stream>>>(eW2, eW2p, MSGH, HDIM);
    pack_weights_kernel<<<F_NW1, 32, 0, stream>>>(nW1, nW1p, 2 * HDIM, MSGH);
    pack_weights_kernel<<<F_NW2, 32, 0, stream>>>(nW2, nW2p, MSGH, HDIM);

    const int ntile = (E + TILE - 1) / TILE;
    const int egrid = ntile < 3072 ? ntile : 3072;   // ~4 tiles per block
    edge_mlp_scatter_kernel<<<egrid, NTHR, 0, stream>>>(
        h, ei, ea, eW1p, eb1, eW2p, eb2, agg, E);

    node_mlp_ln_kernel<<<(N + TILE - 1) / TILE, NTHR, 0, stream>>>(
        h, agg, nW1p, nb1, nW2p, nb2, lng, lnb, (float*)d_out, N);
}
